// Seq2Seq_74500502716460
// MI455X (gfx1250) — matrix operand order repaired and checked by reading, not compiled, not run on hardware
//
#include <hip/hip_runtime.h>
#include <math.h>

// Problem constants (from reference): V, E, H, B, S, T
#define VQ 50257
#define EQ 256
#define HQ 512
#define GQ 2048   // 4*H
#define BQ 64
#define SQ 128
#define TQ 64

typedef __attribute__((ext_vector_type(16))) __bf16 bf16x16;
typedef __attribute__((ext_vector_type(8)))  float  f32x8;

union FragBF { uint4 u[2]; bf16x16 v; };

__device__ __forceinline__ unsigned short f2bf(float f) {
  unsigned int u = __float_as_uint(f);
  u += 0x7FFFu + ((u >> 16) & 1u);          // round-to-nearest-even
  return (unsigned short)(u >> 16);
}
__device__ __forceinline__ float bf2f(unsigned short s) {
  return __uint_as_float(((unsigned int)s) << 16);
}
__device__ __forceinline__ float sigmoidf_(float x) {
  return 1.0f / (1.0f + __expf(-x));
}

// Async global->LDS copy (CDNA5 TDM-adjacent path, tracked by ASYNCcnt).
// ldsOff: byte offset within wave's LDS aperture (low 32 bits of generic ptr).
__device__ __forceinline__ void async_ld_b128(unsigned ldsOff, const void* gaddr) {
  asm volatile("global_load_async_to_lds_b128 %0, %1, off"
               :: "v"(ldsOff), "v"(gaddr) : "memory");
}
#define WAIT_ASYNC_3() asm volatile("s_wait_asynccnt 3" ::: "memory")
#define WAIT_ASYNC_0() asm volatile("s_wait_asynccnt 0" ::: "memory")

// ---------------------------------------------------------------------------
// Core WMMA tile accumulation: C(16x16,f32) += A[16xK](bf16) * W[NxK]^T(bf16)
// A fragment (ISA 7.12.2, 16-bit A 16x32): lane half 0 -> K{0..7,16..23},
// half 1 -> K{8..15,24..31}; row = lane%16.  Two 16B contiguous loads.
// B fragment (32x16): lane holds weight-row n=lane%16, same K order as A
// (half 0 -> K{0..7,16..23}, half 1 -> K{8..15,24..31}).  Two 16B loads.
// ---------------------------------------------------------------------------
__device__ __forceinline__ void gemm_seg(const unsigned short* __restrict__ A, int lda,
                                         const unsigned short* __restrict__ W, int ldw,
                                         int K, int rowA, int rowW, int half, f32x8& c) {
  const unsigned short* ap = A + (size_t)rowA * lda + half * 8;
  const unsigned short* wp = W + (size_t)rowW * ldw + half * 8;
  for (int kb = 0; kb < K; kb += 32) {
    FragBF a, b;
    a.u[0] = *(const uint4*)(ap);
    a.u[1] = *(const uint4*)(ap + 16);
    b.u[0] = *(const uint4*)(wp);
    b.u[1] = *(const uint4*)(wp + 16);
    c = __builtin_amdgcn_wmma_f32_16x16x32_bf16(false, a.v, false, b.v,
                                                (short)0, c, false, false);
    ap += 32;
    wp += 32;
  }
}

// Generic dual-segment GEMM: C[M,N] = A1@W1^T (+ A2@W2^T) (+ bias), one wave
// per 16x16 tile, f32 output, bf16 inputs.  Used for the small latency-bound
// per-step GEMMs of the recurrence.
__global__ __launch_bounds__(256) void k_gemm_dual(
    const unsigned short* __restrict__ A1, int lda1, int K1,
    const unsigned short* __restrict__ W1, int ldw1,
    const unsigned short* __restrict__ A2, int lda2, int K2,
    const unsigned short* __restrict__ W2, int ldw2,
    const float* __restrict__ bias,
    float* __restrict__ C, int ldc, int N, int nTN, int nTiles) {
  int wave = (int)((blockIdx.x * blockDim.x + threadIdx.x) >> 5);
  if (wave >= nTiles) return;
  int tM = wave / nTN, tN = wave - tM * nTN;
  int lane = threadIdx.x & 31, half = lane >> 4, l16 = lane & 15;
  int rowA = tM * 16 + l16;
  int n = tN * 16 + l16;
  int wrow = (n < N) ? n : (N - 1);
  f32x8 c = {};
  gemm_seg(A1, lda1, W1, ldw1, K1, rowA, wrow, half, c);
  if (A2) gemm_seg(A2, lda2, W2, ldw2, K2, rowA, wrow, half, c);
  if (n < N) {
    float bv = bias ? bias[n] : 0.0f;
    int mb = tM * 16 + half * 8;
#pragma unroll
    for (int i = 0; i < 8; ++i)
      C[(size_t)(mb + i) * ldc + n] = c[i] + bv;
  }
}

// ---------------------------------------------------------------------------
// FC head GEMM, LDS-staged + async double-buffered:
//   out[b,t,v] (m=(t-1)*B+b) = fcin[4032,1024] x fc_W[V,1024]^T + fc_b
// Block = 256 threads (8 waves) -> 64x128 output block.  Wave w owns N-tile w
// across 4 M-tiles (4 accumulators, B fragment reused 4x).  Per K-chunk of 32
// the block async-copies A(64x32) and B(128x32) into double-buffered LDS
// (3 x b128 per thread), overlapping the next chunk's copy with WMMA.
// ---------------------------------------------------------------------------
__global__ __launch_bounds__(256) void k_fc_gemm_lds(
    const unsigned short* __restrict__ A, const unsigned short* __restrict__ W,
    const float* __restrict__ bias, float* __restrict__ out, int nBN) {
  __shared__ __align__(16) unsigned short aSh[2][64 * 32];   // 2 x 4KB
  __shared__ __align__(16) unsigned short bSh[2][128 * 32];  // 2 x 8KB

  int bm = blockIdx.x / nBN;
  int bn = blockIdx.x - bm * nBN;
  int tid = threadIdx.x;
  int wave = tid >> 5, lane = tid & 31, half = lane >> 4, l16 = lane & 15;

  // Producer slots: A has 256 b128 slots (row=tid/4, seg=tid%4); B has 512
  // slots, two per thread (rows tid/4 and 64+tid/4).
  int prow = tid >> 2, pseg = tid & 3;
  const unsigned short* agp = A + (size_t)(bm * 64 + prow) * (2 * HQ) + pseg * 8;
  int gr0 = bn * 128 + prow;       if (gr0 >= VQ) gr0 = VQ - 1;
  int gr1 = bn * 128 + 64 + prow;  if (gr1 >= VQ) gr1 = VQ - 1;
  const unsigned short* bgp0 = W + (size_t)gr0 * (2 * HQ) + pseg * 8;
  const unsigned short* bgp1 = W + (size_t)gr1 * (2 * HQ) + pseg * 8;

  f32x8 acc[4] = {};

  auto issue = [&](int buf, int kc) {
    int ko = kc * 32;
    async_ld_b128((unsigned)(size_t)&aSh[buf][prow * 32 + pseg * 8], agp + ko);
    async_ld_b128((unsigned)(size_t)&bSh[buf][prow * 32 + pseg * 8], bgp0 + ko);
    async_ld_b128((unsigned)(size_t)&bSh[buf][(64 + prow) * 32 + pseg * 8], bgp1 + ko);
  };

  issue(0, 0);
  const int NKC = (2 * HQ) / 32;  // 32 chunks
  for (int kc = 0; kc < NKC; ++kc) {
    int cur = kc & 1;
    if (kc + 1 < NKC) {
      issue(cur ^ 1, kc + 1);  // prefetch next chunk into other buffer
      WAIT_ASYNC_3();          // in-order completion => current chunk landed
    } else {
      WAIT_ASYNC_0();
    }
    __syncthreads();           // all waves' async writes visible

    FragBF bf;
    const unsigned short* bs = &bSh[cur][(wave * 16 + l16) * 32 + half * 8];
    bf.u[0] = *(const uint4*)(bs);
    bf.u[1] = *(const uint4*)(bs + 16);
#pragma unroll
    for (int mi = 0; mi < 4; ++mi) {
      FragBF af;
      const unsigned short* as = &aSh[cur][(mi * 16 + l16) * 32 + half * 8];
      af.u[0] = *(const uint4*)(as);
      af.u[1] = *(const uint4*)(as + 16);
      acc[mi] = __builtin_amdgcn_wmma_f32_16x16x32_bf16(false, af.v, false, bf.v,
                                                        (short)0, acc[mi], false, false);
    }
    __syncthreads();           // done reading before next issue overwrites
  }

  int n = bn * 128 + wave * 16 + l16;
  if (n < VQ) {
    float bv = bias[n];
#pragma unroll
    for (int mi = 0; mi < 4; ++mi) {
      int mb = bm * 64 + mi * 16 + half * 8;
#pragma unroll
      for (int i = 0; i < 8; ++i) {
        int m = mb + i;
        int b = m & (BQ - 1);
        int t = (m >> 6) + 1;
        out[(size_t)b * ((size_t)TQ * VQ) + (size_t)t * VQ + n] = acc[mi][i] + bv;
      }
    }
  }
}

// Embedding gather + bf16 convert: out[t][b][e] = emb[idx[b*scols+t]][e]
__global__ void k_embed(const int* __restrict__ idx, int scols,
                        const float* __restrict__ emb,
                        unsigned short* __restrict__ outp, int total) {
  int g = blockIdx.x * blockDim.x + threadIdx.x;
  if (g >= total) return;
  int e = g & (EQ - 1);
  int r = g >> 8;
  int b = r & (BQ - 1);
  int t = r >> 6;
  int tok = idx[b * scols + t];
  outp[g] = f2bf(emb[(size_t)tok * EQ + e]);
}

// LSTM pointwise: gate order i,f,g,o; updates c (f32) and h (bf16 shadow),
// optionally mirrors h into another bf16 buffer (enc_out slice / fcin row).
__global__ void k_lstm_update(const float* __restrict__ gates,
                              float* __restrict__ c,
                              unsigned short* __restrict__ hbf,
                              unsigned short* __restrict__ extra, int extra_ld) {
  int idx = blockIdx.x * blockDim.x + threadIdx.x;  // < B*H
  int b = idx >> 9, hh = idx & (HQ - 1);
  const float* g = gates + (size_t)b * GQ;
  float gi = sigmoidf_(g[hh]);
  float gf = sigmoidf_(g[HQ + hh]);
  float gg = tanhf(g[2 * HQ + hh]);
  float go = sigmoidf_(g[3 * HQ + hh]);
  float cn = gf * c[idx] + gi * gg;
  c[idx] = cn;
  float hn = go * tanhf(cn);
  unsigned short hb = f2bf(hn);
  hbf[idx] = hb;
  if (extra) extra[b * extra_ld + hh] = hb;
}

// Attention score: one wave per (b,s): e = sum_h v_w[h]*tanh(hpart+epart)
// (attn_b was folded into epart's GEMM bias)
__global__ void k_attn_score(const float* __restrict__ hpart,
                             const float* __restrict__ epart,
                             const float* __restrict__ v_w,
                             float* __restrict__ scores) {
  int gwave = (int)((blockIdx.x * blockDim.x + threadIdx.x) >> 5);  // 0..8191
  int lane = threadIdx.x & 31;
  int b = gwave >> 7;
  int s = gwave & (SQ - 1);
  const float* hp = hpart + (size_t)b * HQ;
  const float* ep = epart + ((size_t)s * BQ + b) * HQ;
  float acc = 0.0f;
  for (int h = lane; h < HQ; h += 32) acc += v_w[h] * tanhf(hp[h] + ep[h]);
  for (int off = 16; off; off >>= 1) acc += __shfl_xor(acc, off, 32);
  if (lane == 0) scores[b * SQ + s] = acc;
}

// Softmax over S + context accumulation + assemble decoder LSTM input row.
// One block (128 threads) per batch element.
__global__ void k_softmax_ctx(const float* __restrict__ scores,
                              const unsigned short* __restrict__ enc_out,
                              const unsigned short* __restrict__ xdec_t,
                              unsigned short* __restrict__ xfull,
                              unsigned short* __restrict__ fcin_t) {
  __shared__ float sw[SQ];
  __shared__ float red[4];
  int b = blockIdx.x, tid = threadIdx.x;
  float e = scores[b * SQ + tid];
  float m = e;
  for (int off = 16; off; off >>= 1) m = fmaxf(m, __shfl_xor(m, off, 32));
  if ((tid & 31) == 0) red[tid >> 5] = m;
  __syncthreads();
  m = fmaxf(fmaxf(red[0], red[1]), fmaxf(red[2], red[3]));
  __syncthreads();
  float ex = __expf(e - m);
  float ssum = ex;
  for (int off = 16; off; off >>= 1) ssum += __shfl_xor(ssum, off, 32);
  if ((tid & 31) == 0) red[tid >> 5] = ssum;
  __syncthreads();
  float tot = red[0] + red[1] + red[2] + red[3];
  sw[tid] = ex / tot;
  __syncthreads();
#pragma unroll
  for (int j = 0; j < 4; ++j) {
    int hh = tid + j * 128;
    float acc = 0.0f;
    for (int s = 0; s < SQ; ++s)
      acc += sw[s] * bf2f(enc_out[((size_t)s * BQ + b) * HQ + hh]);
    unsigned short cb = f2bf(acc);
    xfull[b * (EQ + HQ) + EQ + hh] = cb;     // LSTM input: [emb, context]
    fcin_t[b * (2 * HQ) + HQ + hh] = cb;     // FC input:   [h, context]
  }
  for (int j = tid; j < EQ; j += 128)
    xfull[b * (EQ + HQ) + j] = xdec_t[b * EQ + j];
}

// f32 -> bf16, 4 elements/thread (all sizes are multiples of 4)
__global__ void k_cvt_bf16_v4(const float4* __restrict__ in,
                              uint2* __restrict__ out, int n4) {
  int i = blockIdx.x * blockDim.x + threadIdx.x;
  if (i >= n4) return;
  float4 f = in[i];
  uint2 o;
  o.x = (unsigned)f2bf(f.x) | ((unsigned)f2bf(f.y) << 16);
  o.y = (unsigned)f2bf(f.z) | ((unsigned)f2bf(f.w) << 16);
  out[i] = o;
}

__global__ void k_addvec(const float* __restrict__ a, const float* __restrict__ b,
                         float* __restrict__ o, int n) {
  int i = blockIdx.x * blockDim.x + threadIdx.x;
  if (i < n) o[i] = a[i] + b[i];
}

__global__ void k_zero_u32(unsigned int* __restrict__ p, int n) {
  int i = blockIdx.x * blockDim.x + threadIdx.x;
  if (i < n) p[i] = 0u;
}

// zero out[:, 0, :]
__global__ void k_zero_t0(float* __restrict__ out, int total) {
  int g = blockIdx.x * blockDim.x + threadIdx.x;
  if (g >= total) return;
  int b = g / VQ;
  int v = g - b * VQ;
  out[(size_t)b * ((size_t)TQ * VQ) + v] = 0.0f;
}

extern "C" void kernel_launch(void* const* d_in, const int* in_sizes, int n_in,
                              void* d_out, int out_size, void* d_ws, size_t ws_size,
                              hipStream_t stream) {
  (void)in_sizes; (void)n_in; (void)out_size; (void)ws_size;
  const int*   src     = (const int*)d_in[0];
  const int*   trg     = (const int*)d_in[1];
  const float* enc_emb = (const float*)d_in[2];
  const float* enc_Wih = (const float*)d_in[3];
  const float* enc_Whh = (const float*)d_in[4];
  const float* enc_bih = (const float*)d_in[5];
  const float* enc_bhh = (const float*)d_in[6];
  const float* dec_emb = (const float*)d_in[7];
  const float* dec_Wih = (const float*)d_in[8];
  const float* dec_Whh = (const float*)d_in[9];
  const float* dec_bih = (const float*)d_in[10];
  const float* dec_bhh = (const float*)d_in[11];
  const float* attn_W  = (const float*)d_in[12];
  const float* attn_b  = (const float*)d_in[13];
  const float* v_w     = (const float*)d_in[14];
  const float* fc_W    = (const float*)d_in[15];
  const float* fc_b    = (const float*)d_in[16];
  float* out = (float*)d_out;

  // ---- workspace carve-up (256B aligned) ----
  char* wsb = (char*)d_ws;
  size_t off = 0;
  auto take = [&](size_t bytes) -> char* {
    char* p = wsb + off;
    off = (off + bytes + 255) & ~(size_t)255;
    return p;
  };
  unsigned short* fcWb  = (unsigned short*)take((size_t)VQ * 2 * HQ * 2); // 98 MB
  unsigned short* eWih  = (unsigned short*)take((size_t)GQ * EQ * 2);
  unsigned short* eWhh  = (unsigned short*)take((size_t)GQ * HQ * 2);
  unsigned short* dWih  = (unsigned short*)take((size_t)GQ * (EQ + HQ) * 2);
  unsigned short* dWhh  = (unsigned short*)take((size_t)GQ * HQ * 2);
  unsigned short* aWb   = (unsigned short*)take((size_t)HQ * 2 * HQ * 2);
  unsigned short* xenc  = (unsigned short*)take((size_t)SQ * BQ * EQ * 2);
  unsigned short* xdec  = (unsigned short*)take((size_t)(TQ - 1) * BQ * EQ * 2);
  unsigned short* encout= (unsigned short*)take((size_t)SQ * BQ * HQ * 2);
  float*          epart = (float*)take((size_t)SQ * BQ * HQ * 4);
  unsigned short* fcin  = (unsigned short*)take((size_t)(TQ - 1) * BQ * 2 * HQ * 2);
  float*          gates = (float*)take((size_t)BQ * GQ * 4);
  float*          cbuf  = (float*)take((size_t)BQ * HQ * 4);
  unsigned short* hbf   = (unsigned short*)take((size_t)BQ * HQ * 2);
  float*          hpart = (float*)take((size_t)BQ * HQ * 4);
  float*          scores= (float*)take((size_t)BQ * SQ * 4);
  unsigned short* xfull = (unsigned short*)take((size_t)BQ * (EQ + HQ) * 2);
  float*          benc  = (float*)take((size_t)GQ * 4);
  float*          bdec  = (float*)take((size_t)GQ * 4);

  const int TB = 256;
  auto nb = [](long n, int tb) { return (int)((n + tb - 1) / tb); };

  // ---- prep: weight conversions, biases, embeddings, state init ----
  k_cvt_bf16_v4<<<nb((long)VQ * 2 * HQ / 4, TB), TB, 0, stream>>>((const float4*)fc_W, (uint2*)fcWb, VQ * 2 * HQ / 4);
  k_cvt_bf16_v4<<<nb((long)GQ * EQ / 4, TB), TB, 0, stream>>>((const float4*)enc_Wih, (uint2*)eWih, GQ * EQ / 4);
  k_cvt_bf16_v4<<<nb((long)GQ * HQ / 4, TB), TB, 0, stream>>>((const float4*)enc_Whh, (uint2*)eWhh, GQ * HQ / 4);
  k_cvt_bf16_v4<<<nb((long)GQ * (EQ + HQ) / 4, TB), TB, 0, stream>>>((const float4*)dec_Wih, (uint2*)dWih, GQ * (EQ + HQ) / 4);
  k_cvt_bf16_v4<<<nb((long)GQ * HQ / 4, TB), TB, 0, stream>>>((const float4*)dec_Whh, (uint2*)dWhh, GQ * HQ / 4);
  k_cvt_bf16_v4<<<nb((long)HQ * 2 * HQ / 4, TB), TB, 0, stream>>>((const float4*)attn_W, (uint2*)aWb, HQ * 2 * HQ / 4);
  k_addvec<<<nb(GQ, TB), TB, 0, stream>>>(enc_bih, enc_bhh, benc, GQ);
  k_addvec<<<nb(GQ, TB), TB, 0, stream>>>(dec_bih, dec_bhh, bdec, GQ);
  k_embed<<<nb((long)SQ * BQ * EQ, TB), TB, 0, stream>>>(src, SQ, enc_emb, xenc, SQ * BQ * EQ);
  k_embed<<<nb((long)(TQ - 1) * BQ * EQ, TB), TB, 0, stream>>>(trg, TQ, dec_emb, xdec, (TQ - 1) * BQ * EQ);
  k_zero_u32<<<nb(BQ * HQ, TB), TB, 0, stream>>>((unsigned int*)cbuf, BQ * HQ);
  k_zero_u32<<<nb(BQ * HQ / 2, TB), TB, 0, stream>>>((unsigned int*)hbf, BQ * HQ / 2);
  k_zero_t0<<<nb((long)BQ * VQ, TB), TB, 0, stream>>>(out, BQ * VQ);

  // ---- encoder: 128 sequential LSTM steps ----
  for (int s = 0; s < SQ; ++s) {
    k_gemm_dual<<<64, TB, 0, stream>>>(xenc + (size_t)s * BQ * EQ, EQ, EQ, eWih, EQ,
                                       hbf, HQ, HQ, eWhh, HQ,
                                       benc, gates, GQ, GQ, GQ / 16,
                                       (BQ / 16) * (GQ / 16));
    k_lstm_update<<<BQ * HQ / TB, TB, 0, stream>>>(gates, cbuf, hbf,
                                                   encout + (size_t)s * BQ * HQ, HQ);
  }

  // ---- hoisted attention term: epart[s,b,:] = enc_out@attn_W[:,H:]^T + attn_b ----
  k_gemm_dual<<<(SQ * BQ / 16) * (HQ / 16) / 8, TB, 0, stream>>>(
      encout, HQ, HQ, aWb + HQ, 2 * HQ,
      nullptr, 0, 0, nullptr, 0,
      attn_b, epart, HQ, HQ, HQ / 16, (SQ * BQ / 16) * (HQ / 16));

  // ---- decoder: 63 sequential steps with attention ----
  for (int t = 0; t < TQ - 1; ++t) {
    k_gemm_dual<<<16, TB, 0, stream>>>(hbf, HQ, HQ, aWb, 2 * HQ,
                                       nullptr, 0, 0, nullptr, 0,
                                       nullptr, hpart, HQ, HQ, HQ / 16,
                                       (BQ / 16) * (HQ / 16));
    k_attn_score<<<BQ * SQ * 32 / TB, TB, 0, stream>>>(hpart, epart, v_w, scores);
    k_softmax_ctx<<<BQ, 128, 0, stream>>>(scores, encout,
                                          xdec + (size_t)t * BQ * EQ, xfull,
                                          fcin + (size_t)t * BQ * 2 * HQ);
    k_gemm_dual<<<64, TB, 0, stream>>>(xfull, EQ + HQ, EQ + HQ, dWih, EQ + HQ,
                                       hbf, HQ, HQ, dWhh, HQ,
                                       bdec, gates, GQ, GQ, GQ / 16,
                                       (BQ / 16) * (GQ / 16));
    k_lstm_update<<<BQ * HQ / TB, TB, 0, stream>>>(gates, cbuf, hbf,
                                                   fcin + (size_t)t * BQ * 2 * HQ,
                                                   2 * HQ);
  }

  // ---- batched FC head: [4032,1024] x [V,1024]^T, LDS-staged WMMA blocks ----
  {
    int nBN = (VQ + 127) / 128;               // 393 N-blocks of 128
    int nBM = (TQ - 1) * BQ / 64;             // 63 M-blocks of 64
    k_fc_gemm_lds<<<nBM * nBN, TB, 0, stream>>>(fcin, fcWb, fc_b, out, nBN);
  }
}
